// Order2GraphAttentionLayer_4784593567909
// MI455X (gfx1250) — matrix operand order repaired and checked by reading, not compiled, not run on hardware
//
#include <hip/hip_runtime.h>
#include <hip/hip_bf16.h>
#include <stdint.h>

// Problem constants (reference): B=8, N=2048, F_IN=512, F_OUT=256, E=5
#define GB   8
#define GN   2048
#define GFIN 512
#define GD   256

typedef __attribute__((ext_vector_type(16))) __bf16 v16bf;
typedef __attribute__((ext_vector_type(8)))  float  v8f;

#define DEVINL __device__ __forceinline__

DEVINL unsigned short f2bu(float f) {
  unsigned int u = __builtin_bit_cast(unsigned int, f);
  unsigned int r = (u + 0x7FFFu + ((u >> 16) & 1u)) >> 16;
  return (unsigned short)r;
}
DEVINL float bu2f(unsigned short s) {
  unsigned int u = ((unsigned int)s) << 16;
  return __builtin_bit_cast(float, u);
}
DEVINL __bf16 bu2b(unsigned short s) { return __builtin_bit_cast(__bf16, s); }

DEVINL v8f zerov8() {
  v8f z = {0.f, 0.f, 0.f, 0.f, 0.f, 0.f, 0.f, 0.f};
  return z;
}

// Cross-lane reductions via ds_swizzle (and_mask=0x1f; xor<16 stays within the
// 16-lane half that owns one matrix row in the WMMA C/D layout).
template <int XORMASK>
DEVINL float swz(float v) {
  int i = __builtin_amdgcn_ds_swizzle(__builtin_bit_cast(int, v),
                                      (0x1f << 10) | XORMASK);
  return __builtin_bit_cast(float, i);
}
DEVINL float rmax16(float v) {
  v = fmaxf(v, swz<1>(v)); v = fmaxf(v, swz<2>(v));
  v = fmaxf(v, swz<4>(v)); v = fmaxf(v, swz<8>(v));
  return v;
}
DEVINL float rsum16(float v) {
  v += swz<1>(v); v += swz<2>(v); v += swz<4>(v); v += swz<8>(v);
  return v;
}
DEVINL float xhalf(float v) { return swz<16>(v); }

// --------------------------------------------------------------------------
// Fragment-native packed tiles: tile = [32 lanes][16 bf16], 1 KB, 32B/lane.
// A fragment load is one aligned 32-byte vector load (2x global_load_b128).
DEVINL v16bf fragLoad(const unsigned short* tileBase) {
  int l = threadIdx.x & 31;
  return *(const v16bf*)(tileBase + ((size_t)l << 4));
}

// A-fragment built from row-major LDS (per-lane one contiguous 32B chunk).
// A 16x32: lane l -> M=l&15, kh=l>>4; a[2v+i] <-> K=kh*16+v*2+i
DEVINL v16bf fragA_lds(const unsigned short* s, int ld, int k0) {
  int l = threadIdx.x & 31;
  int m = l & 15, kh = (l >> 4) & 1;
  v16bf a;
#pragma unroll
  for (int v = 0; v < 8; ++v) {
    int k = k0 + (kh << 4) + (v << 1);
    const unsigned short* p = s + m * ld + k;
    a[2 * v]     = bu2b(p[0]);
    a[2 * v + 1] = bu2b(p[1]);
  }
  return a;
}

DEVINL v8f wmma_bf16(v16bf a, v16bf b, v8f c) {
  return __builtin_amdgcn_wmma_f32_16x16x32_bf16(false, a, false, b,
                                                 (short)0, c, false, false);
}

// =========================== pack kernels =================================

// Pack f32 row-major [K][N] weight into B-fragment tiles (32x16 K x N).
// B layout: lane=(n&15)+16*((k&31)>>4), e=(k&31)&15, tile=(k>>5)*(N/16)+(n>>4)
__global__ void k_packB(const float* __restrict__ in,
                        unsigned short* __restrict__ out, int K, int N) {
  int i = blockIdx.x * blockDim.x + threadIdx.x;
  if (i >= K * N) return;
  int k = i / N, n = i % N;
  int kl = k & 31;
  int lane = (n & 15) + ((kl >> 4) << 4);
  int e = kl & 15;
  size_t tile = (size_t)(k >> 5) * (N >> 4) + (n >> 4);
  out[(tile * 32 + lane) * 16 + e] = f2bu(in[i]);
}

// Pack feat f32 [B][N][F_IN] into V B-fragment tiles: K=j (32/tile), N=feature.
__global__ void k_packV(const float* __restrict__ in,
                        unsigned short* __restrict__ out) {
  size_t i = (size_t)blockIdx.x * blockDim.x + threadIdx.x;
  if (i >= (size_t)GB * GN * GFIN) return;
  int f = (int)(i % GFIN);
  size_t row = i / GFIN;
  int b = (int)(row >> 11);
  int j = (int)(row & (GN - 1));
  int jl = j & 31;
  int lane = (f & 15) + ((jl >> 4) << 4);
  int e = jl & 15;
  size_t tile = ((size_t)b * (GN / 32) + (j >> 5)) * (GFIN / 16) + (f >> 4);
  out[(tile * 32 + lane) * 16 + e] = f2bu(in[i]);
}

// =========================== k_proj =======================================
// One wave per 16-row tile. Produces hpack (B^T frag layout), gpack (A frag
// layout), Ax, Ay, e1, and out = feat@W1 (h_1).
__global__ __launch_bounds__(32)
void k_proj(const float* __restrict__ feat,
            const float* __restrict__ a1, const float* __restrict__ a2,
            const float* __restrict__ Lw,
            const unsigned short* __restrict__ Wp,
            const unsigned short* __restrict__ W1p,
            const unsigned short* __restrict__ a12p,
            unsigned short* __restrict__ hpack,
            unsigned short* __restrict__ gpack,
            float* __restrict__ Axw, float* __restrict__ Ayw,
            float* __restrict__ e1w, float* __restrict__ out) {
  __shared__ unsigned short s_feat[16 * GFIN];  // 16 KB
  __shared__ unsigned short s_h[16 * GD];       //  8 KB

  int tile = blockIdx.x;                 // b*128 + row_tile
  int b = tile >> 7;
  int i0 = (tile & 127) << 4;
  int lane = threadIdx.x;
  int hl = (lane >> 4) & 1, nl = lane & 15;
  size_t rowbase = (size_t)b * GN + i0;

  // stage feat tile, converting f32 -> bf16 on the fly (coalesced)
  {
    const float4* src = (const float4*)(feat + rowbase * GFIN);
    ushort4* dst = (ushort4*)s_feat;
    for (int i = lane; i < 16 * GFIN / 4; i += 32) {
      float4 v = src[i];
      ushort4 u;
      u.x = f2bu(v.x); u.y = f2bu(v.y); u.z = f2bu(v.z); u.w = f2bu(v.w);
      dst[i] = u;
    }
  }
  __syncthreads();

  v16bf af[16];
#pragma unroll
  for (int kb = 0; kb < 16; ++kb) af[kb] = fragA_lds(s_feat, GFIN, kb * 32);

  // h = feat @ W ; Ax/Ay reductions ; scatter h into B^T fragment pack
  float axp[8], ayp[8];
#pragma unroll
  for (int r = 0; r < 8; ++r) { axp[r] = 0.f; ayp[r] = 0.f; }
#pragma unroll
  for (int nt = 0; nt < 16; ++nt) {
    v8f acc = zerov8();
#pragma unroll
    for (int kb = 0; kb < 16; ++kb)
      acc = wmma_bf16(af[kb], fragLoad(Wp + (size_t)(kb * 16 + nt) * 512), acc);
    int n = nt * 16 + nl;
    float a1v = a1[n], a2v = a2[n];
#pragma unroll
    for (int r = 0; r < 8; ++r) {
      float hv = acc[r];
      axp[r] += hv * a1v;
      ayp[r] += hv * a2v;
      int row = r + 8 * hl;
      unsigned short hbu = f2bu(hv);
      // hpack: B^T layout for S = g·h^T  (lane = row + 16*((n>>4)&1), e = n&15)
      hpack[((size_t)tile * 8 + (nt >> 1)) * 512 +
            (size_t)(row + 16 * (nt & 1)) * 16 + nl] = hbu;
      s_h[row * GD + n] = hbu;
    }
  }
#pragma unroll
  for (int r = 0; r < 8; ++r) {
    float ax = rsum16(axp[r]);
    float ay = rsum16(ayp[r]);
    if (nl == 0) {
      Axw[rowbase + r + 8 * hl] = ax;
      Ayw[rowbase + r + 8 * hl] = ay;
    }
  }
  __syncthreads();

  // g = h @ a12 -> A-fragment pack
  v16bf ha[8];
#pragma unroll
  for (int kb = 0; kb < 8; ++kb) ha[kb] = fragA_lds(s_h, GD, kb * 32);
#pragma unroll
  for (int nt = 0; nt < 16; ++nt) {
    v8f acc = zerov8();
#pragma unroll
    for (int kb = 0; kb < 8; ++kb)
      acc = wmma_bf16(ha[kb], fragLoad(a12p + (size_t)(kb * 16 + nt) * 512), acc);
#pragma unroll
    for (int r = 0; r < 8; ++r) {
      int row = r + 8 * hl;
      // A layout: lane = row + 16*((n>>4)&1), e = n&15
      gpack[((size_t)tile * 8 + (nt >> 1)) * 512 +
            (size_t)(row + 16 * (nt & 1)) * 16 + nl] = f2bu(acc[r]);
    }
  }

  // h_1 = feat @ W1  -> initialize output
#pragma unroll
  for (int nt = 0; nt < 16; ++nt) {
    v8f acc = zerov8();
#pragma unroll
    for (int kb = 0; kb < 16; ++kb)
      acc = wmma_bf16(af[kb], fragLoad(W1p + (size_t)(kb * 16 + nt) * 512), acc);
    int n = nt * 16 + nl;
#pragma unroll
    for (int r = 0; r < 8; ++r)
      out[(rowbase + r + 8 * hl) * GD + n] = acc[r];
  }

  // e1 = feat @ Lw^T  (E=5, scalar path; K split over lane halves)
  {
    int row = nl;
#pragma unroll
    for (int e = 0; e < 5; ++e) {
      float s = 0.f;
      const float* lw = Lw + e * GFIN;
      for (int k = 0; k < GFIN / 2; ++k) {
        int kk = hl * (GFIN / 2) + k;
        s += bu2f(s_feat[row * GFIN + kk]) * lw[kk];
      }
      s += xhalf(s);
      if (hl == 0) e1w[(rowbase + row) * 8 + e] = s;
    }
  }
}

// =========================== k_attn =======================================
// Flash attention + update: one wave owns 16 rows, streams 32 columns/step,
// online softmax, accumulates P@feat into 16x512 f32 registers. All matrix
// operands stream as pre-packed fragments directly from global (L2-resident).
__global__ __launch_bounds__(32)
void k_attn(const unsigned short* __restrict__ featpack,
            const unsigned short* __restrict__ hpack,
            const unsigned short* __restrict__ gpack,
            const float* __restrict__ Axw, const float* __restrict__ Ayw,
            const int* __restrict__ adj,
            const unsigned short* __restrict__ W2p,
            const float* __restrict__ Rw, const float* __restrict__ Bw,
            const float* __restrict__ e1w, float* __restrict__ out) {
  __shared__ unsigned short s_p[16 * 32];      // 1 KB   P bounce
  __shared__ float          s_e2[16 * 8];      // 0.5 KB
  __shared__ unsigned short s_fa[16 * GFIN];   // 16 KB  feat_agg bounce

  int tile = blockIdx.x;
  int b = tile >> 7;
  int i0 = (tile & 127) << 4;
  int lane = threadIdx.x;
  int hl = (lane >> 4) & 1, nl = lane & 15;
  size_t rowbase = (size_t)b * GN + i0;
  size_t adjbase = rowbase * GN;

  v16bf gA[8];
#pragma unroll
  for (int kb = 0; kb < 8; ++kb)
    gA[kb] = fragLoad(gpack + ((size_t)tile * 8 + kb) * 512);

  float ayv[8], m[8], ssum[8];
#pragma unroll
  for (int r = 0; r < 8; ++r) {
    ayv[r] = Ayw[rowbase + r + 8 * hl];
    m[r] = -3.0e38f;
    ssum[r] = 0.f;
  }
  v8f acc[32];
#pragma unroll
  for (int nt = 0; nt < 32; ++nt) acc[nt] = zerov8();

  for (int jb = 0; jb < GN; jb += 32) {
    // two 16x16 score tiles: S = g_i @ h_j^T (K=256), operands pre-packed
    v8f st[2];
#pragma unroll
    for (int t = 0; t < 2; ++t) {
      size_t ht = (size_t)(b * 128 + (jb >> 4) + t) * 8;
      v8f s = zerov8();
#pragma unroll
      for (int kb = 0; kb < 8; ++kb)
        s = wmma_bf16(gA[kb], fragLoad(hpack + (ht + kb) * 512), s);
      st[t] = s;
    }

    // logits + leaky relu + adjacency mask
    float ev[2][8];
#pragma unroll
    for (int t = 0; t < 2; ++t) {
      int j = jb + t * 16 + nl;
      float axv = Axw[(size_t)b * GN + j];
#pragma unroll
      for (int r = 0; r < 8; ++r) {
        float lg = st[t][r] + axv + ayv[r];
        float e = lg > 0.f ? lg : 0.2f * lg;
        int a = adj[adjbase + (size_t)(r + 8 * hl) * GN + j];
        ev[t][r] = (a > 0) ? e : -9.0e15f;
      }
    }

    // online softmax (row stats replicated across each 16-lane half)
    float scs[8];
#pragma unroll
    for (int r = 0; r < 8; ++r) {
      float rm = rmax16(fmaxf(ev[0][r], ev[1][r]));
      float mn = fmaxf(m[r], rm);
      float sc = __expf(m[r] - mn);
      float p0 = __expf(ev[0][r] - mn);
      float p1 = __expf(ev[1][r] - mn);
      ssum[r] = ssum[r] * sc + rsum16(p0 + p1);
      m[r] = mn;
      scs[r] = sc;
      s_p[(r + 8 * hl) * 32 + nl] = f2bu(p0);
      s_p[(r + 8 * hl) * 32 + 16 + nl] = f2bu(p1);
    }
    __syncthreads();

    // rescale running accumulator
#pragma unroll
    for (int nt = 0; nt < 32; ++nt) {
#pragma unroll
      for (int r = 0; r < 8; ++r) acc[nt][r] *= scs[r];
    }

    // acc += P @ V ; V fragments stream directly from packed global (L2)
    v16bf pA = fragA_lds(s_p, 32, 0);
    size_t vt = ((size_t)b * (GN / 32) + (jb >> 5)) * 32;
#pragma unroll
    for (int nt = 0; nt < 32; ++nt)
      acc[nt] = wmma_bf16(pA, fragLoad(featpack + (vt + nt) * 512), acc[nt]);
    __syncthreads();
  }

  // normalize -> feat_agg, bounce through LDS as bf16 (16x512)
  float rinv[8];
#pragma unroll
  for (int r = 0; r < 8; ++r) rinv[r] = 1.0f / ssum[r];
#pragma unroll
  for (int nt = 0; nt < 32; ++nt) {
    int n = nt * 16 + nl;
#pragma unroll
    for (int r = 0; r < 8; ++r)
      s_fa[(r + 8 * hl) * GFIN + n] = f2bu(acc[nt][r] * rinv[r]);
  }
  __syncthreads();

  // e2 = feat_agg @ Rw^T
  {
    int row = nl;
#pragma unroll
    for (int e = 0; e < 5; ++e) {
      float s = 0.f;
      const float* rw = Rw + e * GFIN;
      for (int k = 0; k < GFIN / 2; ++k) {
        int kk = hl * (GFIN / 2) + k;
        s += bu2f(s_fa[row * GFIN + kk]) * rw[kk];
      }
      s += xhalf(s);
      if (hl == 0) s_e2[row * 8 + e] = s;
    }
  }
  __syncthreads();

  float e1r[8][5], e2r[8][5];
#pragma unroll
  for (int r = 0; r < 8; ++r) {
    int row = r + 8 * hl;
#pragma unroll
    for (int e = 0; e < 5; ++e) {
      e1r[r][e] = e1w[(rowbase + row) * 8 + e];
      e2r[r][e] = s_e2[row * 8 + e];
    }
  }

  // h_2 = feat_agg @ W2  (K=512)
  v8f acc2[16];
#pragma unroll
  for (int nt = 0; nt < 16; ++nt) acc2[nt] = zerov8();
#pragma unroll
  for (int kb = 0; kb < 16; ++kb) {
    v16bf faA = fragA_lds(s_fa, GFIN, kb * 32);
#pragma unroll
    for (int nt = 0; nt < 16; ++nt)
      acc2[nt] = wmma_bf16(faA, fragLoad(W2p + (size_t)(kb * 16 + nt) * 512),
                           acc2[nt]);
  }

  // out += h_2 + bilinear(e1, B_w, e2)
#pragma unroll
  for (int nt = 0; nt < 16; ++nt) {
    int o = nt * 16 + nl;
    const float* bw = Bw + (size_t)o * 25;
#pragma unroll
    for (int r = 0; r < 8; ++r) {
      float h12 = 0.f;
#pragma unroll
      for (int p = 0; p < 5; ++p) {
#pragma unroll
        for (int q = 0; q < 5; ++q)
          h12 += bw[p * 5 + q] * e1r[r][p] * e2r[r][q];
      }
      size_t oi = (rowbase + r + 8 * hl) * GD + o;
      out[oi] += acc2[nt][r] + h12;
    }
  }
}

// =========================== launch =======================================

extern "C" void kernel_launch(void* const* d_in, const int* in_sizes, int n_in,
                              void* d_out, int out_size, void* d_ws,
                              size_t ws_size, hipStream_t stream) {
  (void)in_sizes; (void)n_in; (void)out_size; (void)ws_size;
  const float* feat = (const float*)d_in[0];
  const int*   adj  = (const int*)d_in[1];
  const float* W    = (const float*)d_in[2];
  const float* W1   = (const float*)d_in[3];
  const float* W2   = (const float*)d_in[4];
  const float* a1   = (const float*)d_in[5];
  const float* a2   = (const float*)d_in[6];
  const float* a12  = (const float*)d_in[7];
  const float* Lw   = (const float*)d_in[8];
  const float* Rw   = (const float*)d_in[9];
  const float* Bw   = (const float*)d_in[10];
  float* out = (float*)d_out;

  char* ws = (char*)d_ws;
  size_t off = 0;
  unsigned short* featpack = (unsigned short*)(ws + off); off += (size_t)GB * GN * GFIN * 2;
  unsigned short* hpack    = (unsigned short*)(ws + off); off += (size_t)GB * GN * GD * 2;
  unsigned short* gpack    = (unsigned short*)(ws + off); off += (size_t)GB * GN * GD * 2;
  unsigned short* Wp       = (unsigned short*)(ws + off); off += (size_t)GFIN * GD * 2;
  unsigned short* W1p      = (unsigned short*)(ws + off); off += (size_t)GFIN * GD * 2;
  unsigned short* W2p      = (unsigned short*)(ws + off); off += (size_t)GFIN * GD * 2;
  unsigned short* a12p     = (unsigned short*)(ws + off); off += (size_t)GD * GD * 2;
  float* Axw = (float*)(ws + off); off += (size_t)GB * GN * 4;
  float* Ayw = (float*)(ws + off); off += (size_t)GB * GN * 4;
  float* e1w = (float*)(ws + off); off += (size_t)GB * GN * 8 * 4;

  k_packV<<<(GB * GN * GFIN + 255) / 256, 256, 0, stream>>>(feat, featpack);
  k_packB<<<(GFIN * GD + 255) / 256, 256, 0, stream>>>(W,   Wp,   GFIN, GD);
  k_packB<<<(GFIN * GD + 255) / 256, 256, 0, stream>>>(W1,  W1p,  GFIN, GD);
  k_packB<<<(GFIN * GD + 255) / 256, 256, 0, stream>>>(W2,  W2p,  GFIN, GD);
  k_packB<<<(GD * GD + 255) / 256, 256, 0, stream>>>(a12, a12p, GD, GD);

  int tiles = GB * GN / 16;  // 1024
  k_proj<<<tiles, 32, 0, stream>>>(feat, a1, a2, Lw, Wp, W1p, a12p,
                                   hpack, gpack, Axw, Ayw, e1w, out);
  k_attn<<<tiles, 32, 0, stream>>>(featpack, hpack, gpack, Axw, Ayw, adj,
                                   W2p, Rw, Bw, e1w, out);
}
